// renderer_31542239822155
// MI455X (gfx1250) — matrix operand order repaired and checked by reading, not compiled, not run on hardware
//
#include <hip/hip_runtime.h>
#include <hip/hip_fp16.h>

typedef __attribute__((ext_vector_type(16))) _Float16 v16h;
typedef __attribute__((ext_vector_type(8)))  _Float16 v8h;
typedef __attribute__((ext_vector_type(8)))  float    v8f;

#define N_G   2048
#define D_F   64
#define IMG_H 128
#define IMG_W 128
#define NEARP 0.01f
#define RCLIP 3.0f
#define BLURC 0.3f
#define A_MIN (1.0f/255.0f)
#define A_MAX 0.999f
#define CHUNK 32
#define NCHUNK (N_G/CHUNK)   // 64
#define NTILE  (D_F/16)      // 4

// ---------------------------------------------------------------- preprocess
__global__ __launch_bounds__(256)
void gs_preprocess(const float* __restrict__ means, const float* __restrict__ quats,
                   const float* __restrict__ scales, const float* __restrict__ opl,
                   const float* __restrict__ c2o, const float* __restrict__ Kmat,
                   float* __restrict__ mx, float* __restrict__ my,
                   float* __restrict__ cA, float* __restrict__ cB, float* __restrict__ cC,
                   float* __restrict__ op, float* __restrict__ zk)
{
    int n = blockIdx.x * blockDim.x + threadIdx.x;
    if (n >= N_G) return;

    // viewmat = inv(c2o), rigid: Rv = Rc^T, tv = -Rc^T * tc
    float R[3][3], tv[3];
    float tc0 = c2o[3], tc1 = c2o[7], tc2 = c2o[11];
    #pragma unroll
    for (int i = 0; i < 3; ++i)
        #pragma unroll
        for (int j = 0; j < 3; ++j) R[i][j] = c2o[j*4 + i];
    #pragma unroll
    for (int i = 0; i < 3; ++i) tv[i] = -(R[i][0]*tc0 + R[i][1]*tc1 + R[i][2]*tc2);

    float fx = Kmat[0], fy = Kmat[4], cx = Kmat[2], cy = Kmat[5];

    float m0 = means[n*3+0], m1 = means[n*3+1], m2 = means[n*3+2];
    float x = R[0][0]*m0 + R[0][1]*m1 + R[0][2]*m2 + tv[0];
    float y = R[1][0]*m0 + R[1][1]*m1 + R[1][2]*m2 + tv[1];
    float z = R[2][0]*m0 + R[2][1]*m1 + R[2][2]*m2 + tv[2];

    // quat -> rotation
    float qw = quats[n*4+0], qx = quats[n*4+1], qy = quats[n*4+2], qz = quats[n*4+3];
    float qn = rsqrtf(qw*qw + qx*qx + qy*qy + qz*qz);
    qw *= qn; qx *= qn; qy *= qn; qz *= qn;
    float Rq[3][3] = {
        {1.f-2.f*(qy*qy+qz*qz), 2.f*(qx*qy-qw*qz),     2.f*(qx*qz+qw*qy)},
        {2.f*(qx*qy+qw*qz),     1.f-2.f*(qx*qx+qz*qz), 2.f*(qy*qz-qw*qx)},
        {2.f*(qx*qz-qw*qy),     2.f*(qy*qz+qw*qx),     1.f-2.f*(qx*qx+qy*qy)}};

    float s0 = __expf(scales[n*3+0]), s1 = __expf(scales[n*3+1]), s2 = __expf(scales[n*3+2]);
    float M[3][3];
    #pragma unroll
    for (int i = 0; i < 3; ++i) { M[i][0] = Rq[i][0]*s0; M[i][1] = Rq[i][1]*s1; M[i][2] = Rq[i][2]*s2; }

    float CW[3][3];
    #pragma unroll
    for (int i = 0; i < 3; ++i)
        #pragma unroll
        for (int j = 0; j < 3; ++j)
            CW[i][j] = M[i][0]*M[j][0] + M[i][1]*M[j][1] + M[i][2]*M[j][2];

    float T1[3][3], CCm[3][3];
    #pragma unroll
    for (int i = 0; i < 3; ++i)
        #pragma unroll
        for (int j = 0; j < 3; ++j)
            T1[i][j] = R[i][0]*CW[0][j] + R[i][1]*CW[1][j] + R[i][2]*CW[2][j];
    #pragma unroll
    for (int i = 0; i < 3; ++i)
        #pragma unroll
        for (int j = 0; j < 3; ++j)
            CCm[i][j] = T1[i][0]*R[j][0] + T1[i][1]*R[j][1] + T1[i][2]*R[j][2];

    float rz  = 1.f / z;
    float J00 = fx*rz, J02 = -fx*x*rz*rz, J11 = fy*rz, J12 = -fy*y*rz*rz;

    float u0 = J00*CCm[0][0] + J02*CCm[2][0];
    float u2 = J00*CCm[0][2] + J02*CCm[2][2];
    float v0 = J11*CCm[1][0] + J12*CCm[2][0];
    float v1 = J11*CCm[1][1] + J12*CCm[2][1];
    float v2 = J11*CCm[1][2] + J12*CCm[2][2];

    float a  = u0*J00 + u2*J02 + BLURC;
    float b  = v0*J00 + v2*J02;
    float cc = v1*J11 + v2*J12 + BLURC;
    float det = a*cc - b*b;

    float lam = 0.5f*(a+cc) + sqrtf(fmaxf(0.25f*(a-cc)*(a-cc) + b*b, 1e-12f));
    float radius = 3.f * sqrtf(lam);
    bool valid = (z > NEARP) && (det > 0.f) && (radius > RCLIP);

    float invdet = 1.f / det;
    float sig_op = 1.f / (1.f + __expf(-opl[n]));

    if (valid) {
        mx[n] = fx*x*rz + cx;
        my[n] = fy*y*rz + cy;
        cA[n] = cc * invdet;
        cB[n] = -b * invdet;
        cC[n] = a  * invdet;
        op[n] = sig_op;
        zk[n] = z;
    } else {
        mx[n] = 0.f; my[n] = 0.f; cA[n] = 0.f; cB[n] = 0.f; cC[n] = 0.f;
        op[n] = 0.f;          // alpha == 0 -> no contribution, T unchanged
        zk[n] = 1e30f;        // sort to the back (contribution is zero anyway)
    }
}

// ---------------------------------------------------------------- bitonic sort
__global__ __launch_bounds__(1024)
void gs_sort(const float* __restrict__ zk, int* __restrict__ order)
{
    __shared__ float sk[N_G];
    __shared__ int   sv[N_G];
    int t = threadIdx.x;
    for (int i = t; i < N_G; i += 1024) { sk[i] = zk[i]; sv[i] = i; }
    __syncthreads();

    for (int k = 2; k <= N_G; k <<= 1) {
        for (int j = k >> 1; j > 0; j >>= 1) {
            for (int i = t; i < N_G; i += 1024) {
                int ixj = i ^ j;
                if (ixj > i) {
                    bool up = ((i & k) == 0);
                    float av = sk[i], bv = sk[ixj];
                    bool sw = up ? (av > bv) : (av < bv);
                    if (sw) {
                        sk[i] = bv; sk[ixj] = av;
                        int tvv = sv[i]; sv[i] = sv[ixj]; sv[ixj] = tvv;
                    }
                }
            }
            __syncthreads();
        }
    }
    for (int i = t; i < N_G; i += 1024) order[i] = sv[i];
}

// ---------------------------------------------------------------- gather params into depth order
__global__ __launch_bounds__(256)
void gs_gather(const int* __restrict__ order,
               const float* __restrict__ mx, const float* __restrict__ my,
               const float* __restrict__ cA, const float* __restrict__ cB,
               const float* __restrict__ cC, const float* __restrict__ op,
               float* __restrict__ smx, float* __restrict__ smy,
               float* __restrict__ scA, float* __restrict__ scB,
               float* __restrict__ scC, float* __restrict__ sop)
{
    int i = blockIdx.x * blockDim.x + threadIdx.x;
    if (i >= N_G) return;
    int g = order[i];
    smx[i] = mx[g]; smy[i] = my[g];
    scA[i] = cA[g]; scB[i] = cB[g]; scC[i] = cC[g];
    sop[i] = op[g];
}

// ---------------------------------------------------------------- pack colors into WMMA-B fragment layout (f16, depth ordered)
// B is KxN = 32x16 per fragment.  B k order used by this code (wave32, contiguous, same as A):
//   lanes 0-15 : N = lane,    element e holds K = e        (0..15)
//   lanes 16-31: N = lane-16, element e holds K = 16 + e   (16..31)
// packB[((chunk*NTILE + tile)*32 + lane)*16 + e]
__global__ __launch_bounds__(256)
void gs_pack(const int* __restrict__ order, const float* __restrict__ colors,
             _Float16* __restrict__ packB)
{
    int f = blockIdx.x * blockDim.x + threadIdx.x;       // 0 .. 131071
    if (f >= NCHUNK * NTILE * 32 * 16) return;
    int e    = f & 15;
    int lane = (f >> 4) & 31;
    int tl   = (f >> 9) & 3;
    int c    = f >> 11;
    int nn   = lane & 15;
    int K    = e + 16 * (lane >> 4);
    int g    = order[c * CHUNK + K];
    packB[f] = (_Float16)colors[g * D_F + tl * 16 + nn];
}

// ---------------------------------------------------------------- rasterize: alpha/cumprod on VALU, feature reduction on WMMA
__global__ __launch_bounds__(256)
void gs_rasterize(const float* __restrict__ smx, const float* __restrict__ smy,
                  const float* __restrict__ scA, const float* __restrict__ scB,
                  const float* __restrict__ scC, const float* __restrict__ sop,
                  const _Float16* __restrict__ packB, float* __restrict__ out)
{
    __shared__ float pmx[CHUNK], pmy[CHUNK], pA[CHUNK], pB[CHUNK], pC[CHUNK], pO[CHUNK];
    __shared__ alignas(16) _Float16 wbuf[256][CHUNK];   // 16 KB: per-pixel weights for one chunk

    int tid  = threadIdx.x;
    int lane = tid & 31;
    int wave = tid >> 5;
    int hi   = lane >> 4;
    int row  = lane & 15;

    int pix = blockIdx.x * 256 + tid;                   // one pixel per thread
    float pxf = (float)(pix % IMG_W) + 0.5f;
    float pyf = (float)(pix / IMG_W) + 0.5f;

    float T = 1.0f;                                     // running transmittance
    v8f acc[2][NTILE];
    #pragma unroll
    for (int m = 0; m < 2; ++m)
        #pragma unroll
        for (int t = 0; t < NTILE; ++t)
            acc[m][t] = (v8f){0.f,0.f,0.f,0.f,0.f,0.f,0.f,0.f};

    for (int c = 0; c < NCHUNK; ++c) {
        int k0 = c * CHUNK;
        if (tid < CHUNK) {
            pmx[tid] = smx[k0 + tid]; pmy[tid] = smy[k0 + tid];
            pA[tid]  = scA[k0 + tid]; pB[tid]  = scB[k0 + tid];
            pC[tid]  = scC[k0 + tid]; pO[tid]  = sop[k0 + tid];
        }
        __syncthreads();

        // sequential (depth-ordered) alpha + transmittance for this pixel
        #pragma unroll 8
        for (int k = 0; k < CHUNK; ++k) {
            float dx = pxf - pmx[k];
            float dy = pyf - pmy[k];
            float sg = 0.5f * (pA[k]*dx*dx + pC[k]*dy*dy) + pB[k]*dx*dy;
            float al = fminf(pO[k] * __expf(-sg), A_MAX);
            al = (sg >= 0.f && al >= A_MIN) ? al : 0.f;
            wbuf[tid][k] = (_Float16)(al * T);
            T *= (1.f - al);
        }
        __syncthreads();

        // A fragments (16-bit A 16x32 layout, K order matches packB):
        //   lanes 0-15 : M=lane,    element e holds K = e        (0..15)
        //   lanes 16-31: M=lane-16, element e holds K = 16 + e   (16..31)
        v16h afrag[2];
        #pragma unroll
        for (int m = 0; m < 2; ++m) {
            const v8h* rp = (const v8h*)&wbuf[wave*32 + m*16 + row][0];
            v8h lo = rp[2 * hi];        // halves [16*hi .. 16*hi+7]
            v8h up = rp[2 * hi + 1];    // halves [16*hi+8 .. 16*hi+15]
            v16h a;
            #pragma unroll
            for (int e = 0; e < 8; ++e) { a[e] = lo[e]; a[8 + e] = up[e]; }
            afrag[m] = a;
        }

        #pragma unroll
        for (int t = 0; t < NTILE; ++t) {
            v16h bfrag = *(const v16h*)(packB + ((size_t)(c * NTILE + t) * 32 + lane) * 16);
            #pragma unroll
            for (int m = 0; m < 2; ++m) {
                acc[m][t] = __builtin_amdgcn_wmma_f32_16x16x32_f16(
                    false, afrag[m], false, bfrag, (short)0, acc[m][t], false, false);
            }
        }
    }

    // C/D 16x16 f32 layout: lanes 0-15: M=r, N=lane; lanes 16-31: M=8+r, N=lane-16
    int base = blockIdx.x * 256 + wave * 32;
    #pragma unroll
    for (int m = 0; m < 2; ++m)
        #pragma unroll
        for (int t = 0; t < NTILE; ++t)
            #pragma unroll
            for (int r = 0; r < 8; ++r) {
                int prow = base + m*16 + r + 8*hi;
                out[(size_t)prow * D_F + t*16 + (lane & 15)] = acc[m][t][r];
            }
}

// ---------------------------------------------------------------- launch
extern "C" void kernel_launch(void* const* d_in, const int* in_sizes, int n_in,
                              void* d_out, int out_size, void* d_ws, size_t ws_size,
                              hipStream_t stream)
{
    const float* means  = (const float*)d_in[0];
    const float* quats  = (const float*)d_in[1];
    const float* scales = (const float*)d_in[2];
    const float* opac   = (const float*)d_in[3];
    const float* colors = (const float*)d_in[4];
    const float* c2o    = (const float*)d_in[5];
    const float* Kmat   = (const float*)d_in[6];
    (void)in_sizes; (void)n_in; (void)out_size; (void)ws_size;

    float* ws = (float*)d_ws;
    float* mx  = ws + 0*N_G;
    float* my  = ws + 1*N_G;
    float* cA  = ws + 2*N_G;
    float* cB  = ws + 3*N_G;
    float* cC  = ws + 4*N_G;
    float* op  = ws + 5*N_G;
    float* zk  = ws + 6*N_G;
    int*   order = (int*)(ws + 7*N_G);
    float* smx = ws + 8*N_G;
    float* smy = ws + 9*N_G;
    float* scA = ws + 10*N_G;
    float* scB = ws + 11*N_G;
    float* scC = ws + 12*N_G;
    float* sop = ws + 13*N_G;
    _Float16* packB = (_Float16*)(ws + 14*N_G);   // 131072 halves = 256 KB

    gs_preprocess<<<N_G/256, 256, 0, stream>>>(means, quats, scales, opac, c2o, Kmat,
                                               mx, my, cA, cB, cC, op, zk);
    gs_sort<<<1, 1024, 0, stream>>>(zk, order);
    gs_gather<<<N_G/256, 256, 0, stream>>>(order, mx, my, cA, cB, cC, op,
                                           smx, smy, scA, scB, scC, sop);
    gs_pack<<<(NCHUNK*NTILE*32*16)/256, 256, 0, stream>>>(order, colors, packB);
    gs_rasterize<<<(IMG_H*IMG_W)/256, 256, 0, stream>>>(smx, smy, scA, scB, scC, sop,
                                                        packB, (float*)d_out);
}
